// CyclicNoiseOscillator_695784701954
// MI455X (gfx1250) — matrix operand order repaired and checked by reading, not compiled, not run on hardware
//
#include <hip/hip_runtime.h>
#include <hip/hip_bf16.h>

typedef __attribute__((ext_vector_type(16))) _Float16 v16h;
typedef __attribute__((ext_vector_type(8)))  float    v8f;
typedef __attribute__((ext_vector_type(4)))  float    v4f;

// Problem constants
#define L_SAMP   360000
#define T_FRAMES 750
#define FRAME    480
#define KSIZE    1003
#define NBATCH   32
#define XSTRIDE  361536       // L + 1536 zero pad (conv window over-read)
#define NCHUNK   75
#define CSAMP    4800         // samples per chunk
#define SCANT    192          // threads per scan block
#define SPT      25           // samples per scan thread (192*25 = 4800)
#define KCHUNKS  32           // 32 k-chunks of 32 => K=1024 (kernel zero padded)
#define QUADS    352          // ceil(ceil(360000/256)/4); 4 tiles (1024 outputs) per wave

// ---------------- helpers ----------------
__device__ __forceinline__ float f0_up_at(const float* __restrict__ f0b, int i) {
    float x = ((float)i + 0.5f) * (1.0f / (float)FRAME) - 0.5f;
    x = fminf(fmaxf(x, 0.0f), (float)(T_FRAMES - 1));
    int i0 = (int)x;                       // floor, x >= 0
    int i1 = min(i0 + 1, T_FRAMES - 1);
    float w = x - (float)i0;
    return f0b[i0] * (1.0f - w) + f0b[i1] * w;
}

// ---------------- Pass A: chunk sums of delta = -f0_up/SR (f64) ----------------
__global__ void cno_chunk_sums(const float* __restrict__ f0, double* __restrict__ partial) {
    int b = blockIdx.y, ch = blockIdx.x, t = threadIdx.x;
    const float* f0b = f0 + b * T_FRAMES;
    int g0 = ch * CSAMP + t * SPT;
    double s = 0.0;
    #pragma unroll
    for (int i = 0; i < SPT; ++i)
        s += -(double)f0_up_at(f0b, g0 + i) * (1.0 / 24000.0);
    __shared__ double red[SCANT];
    red[t] = s;
    __syncthreads();
    if (t < 64) red[t] = red[t] + red[t + 64] + red[t + 128];
    __syncthreads();
    for (int off = 32; off > 0; off >>= 1) {
        if (t < off) red[t] += red[t + off];
        __syncthreads();
    }
    if (t == 0) partial[b * NCHUNK + ch] = red[0];
}

// ---------------- Pass B: per-batch exclusive scan of chunk sums ----------------
__global__ void cno_scan_chunks(const double* __restrict__ partial, double* __restrict__ prefix) {
    int b = threadIdx.x;
    if (b >= NBATCH) return;
    double run = 0.0;
    for (int c = 0; c < NCHUNK; ++c) {
        prefix[b * (NCHUNK + 1) + c] = run;
        run += partial[b * NCHUNK + c];
    }
    prefix[b * (NCHUNK + 1) + NCHUNK] = run;   // total (== rad[L-1])
}

// ---------------- Pass C: sawtooth/impulse/voiced generation ----------------
__global__ void cno_gen_impulse(const float* __restrict__ f0, const double* __restrict__ prefix,
                                _Float16* __restrict__ xbuf, unsigned char* __restrict__ voiced) {
    int b = blockIdx.y, ch = blockIdx.x, t = threadIdx.x;
    const float* f0b = f0 + b * T_FRAMES;
    int g0 = ch * CSAMP + t * SPT;
    float dl[SPT];
    double s = 0.0;
    #pragma unroll
    for (int i = 0; i < SPT; ++i) {
        float fu = f0_up_at(f0b, g0 + i);
        voiced[(size_t)b * L_SAMP + g0 + i] = (fu >= 20.0f) ? 1 : 0;
        float d = -fu * (1.0f / 24000.0f);
        dl[i] = d;
        s += (double)d;
    }
    __shared__ double sc[SCANT];
    sc[t] = s;
    __syncthreads();
    for (int off = 1; off < SCANT; off <<= 1) {
        double v = sc[t];
        double a = (t >= off) ? sc[t - off] : 0.0;
        __syncthreads();
        sc[t] = v + a;
        __syncthreads();
    }
    double excl = sc[t] - s;
    double base = prefix[b * (NCHUNK + 1) + ch] + excl;      // rad[g0-1]
    double total = prefix[b * (NCHUNK + 1) + NCHUNK];        // rad[L-1]
    double rad = base;
    double pv = (g0 == 0) ? total : base;
    double sprev = pv - floor(pv);
    _Float16* xb = xbuf + (size_t)b * XSTRIDE + g0;
    #pragma unroll
    for (int i = 0; i < SPT; ++i) {
        rad += (double)dl[i];
        double saw = rad - floor(rad);
        xb[i] = (_Float16)((float)(saw - sprev));
        sprev = saw;
    }
    if (ch == NCHUNK - 1) {   // zero the conv over-read pad
        for (int k = t; k < XSTRIDE - L_SAMP; k += SCANT)
            xbuf[(size_t)b * XSTRIDE + L_SAMP + k] = (_Float16)0.0f;
    }
}

// ---------------- Build A-matrix (band-Toeplitz kernel) WMMA fragments ----------------
// A[m, j] = w[j - m], j = 32*c + kc, packed in contiguous k order (k = 16*(lane/16) + h).
__global__ void cno_build_A(const float* __restrict__ kern, _Float16* __restrict__ Afrag) {
    int c = blockIdx.x;        // k-chunk
    int lane = threadIdx.x;    // 0..31
    int m = lane & 15;
    bool hi = lane >= 16;
    #pragma unroll
    for (int h = 0; h < 16; ++h) {
        int kc = hi ? h + 16 : h;
        int j = 32 * c + kc - m;
        float v = (j >= 0 && j < KSIZE) ? kern[j] : 0.0f;
        Afrag[((size_t)c * 32 + lane) * 16 + h] = (_Float16)v;
    }
}

// ---------------- WMMA FIR + branchless voiced-mix epilogue ----------------
__device__ __forceinline__ void cno_epilogue(const v8f& acc, int i0, int lane, int b,
                                             const unsigned char* __restrict__ voiced,
                                             const float* __restrict__ noise,
                                             float* __restrict__ out) {
    // f32 16x16 C/D layout: lane<16 -> N=lane, M=r; lane>=16 -> N=lane-16, M=8+r
    int ob = i0 + 16 * (lane & 15) + ((lane & 16) ? 8 : 0);
    if (ob >= L_SAMP) return;                      // L and ob are multiples of 8
    size_t o = (size_t)b * L_SAMP + ob;
    uint2 vv = *(const uint2*)(voiced + o);        // 8 voiced bytes, branchless selects below
    const v4f* np = (const v4f*)(noise + o);
    v4f n0 = __builtin_nontemporal_load(np);
    v4f n1 = __builtin_nontemporal_load(np + 1);
    v4f o0, o1;
    o0.x = (vv.x & 0x000000FFu) ? acc[0] : n0.x;
    o0.y = (vv.x & 0x0000FF00u) ? acc[1] : n0.y;
    o0.z = (vv.x & 0x00FF0000u) ? acc[2] : n0.z;
    o0.w = (vv.x & 0xFF000000u) ? acc[3] : n0.w;
    o1.x = (vv.y & 0x000000FFu) ? acc[4] : n1.x;
    o1.y = (vv.y & 0x0000FF00u) ? acc[5] : n1.y;
    o1.z = (vv.y & 0x00FF0000u) ? acc[6] : n1.z;
    o1.w = (vv.y & 0xFF000000u) ? acc[7] : n1.w;
    v4f* op = (v4f*)(out + o);
    __builtin_nontemporal_store(o0, op);
    __builtin_nontemporal_store(o1, op + 1);
}

__global__ void __launch_bounds__(128) cno_fir_wmma(
        const _Float16* __restrict__ xbuf, const _Float16* __restrict__ Afrag,
        const unsigned char* __restrict__ voiced, const float* __restrict__ noise,
        float* __restrict__ out) {
    __shared__ __align__(16) _Float16 smA[KCHUNKS * 32 * 16];   // 32 KB
    int b = blockIdx.y;
    int tid = threadIdx.x;
    {   // cooperative fill of the shared Toeplitz fragments (reused by every tile)
        const uint4* src = (const uint4*)Afrag;
        uint4* dst = (uint4*)smA;
        #pragma unroll 4
        for (int k = tid; k < (KCHUNKS * 32 * 16) / 8; k += 128) dst[k] = src[k];
    }
    __syncthreads();
    int wave = tid >> 5, lane = tid & 31;
    int quad = blockIdx.x * 4 + wave;              // 88*4 == QUADS exactly
    int i0 = quad * 1024;                          // four 256-output tiles
    // B in contiguous k order, as A: lane<16 -> N=lane, K=0..15 ; lane>=16 -> N=lane-16, K=16..31
    const _Float16* xb = xbuf + (size_t)b * XSTRIDE + i0 + 16 * (lane & 15) + (lane & 16);
    v8f acc0 = {};
    v8f acc1 = {};
    v8f acc2 = {};
    v8f acc3 = {};
    #pragma unroll 2
    for (int c = 0; c < KCHUNKS; ++c) {
        v16h a  = *(const v16h*)(smA + ((size_t)c * 32 + lane) * 16);
        v16h b0 = *(const v16h*)(xb + 32 * c);
        v16h b1 = *(const v16h*)(xb + 256 + 32 * c);
        v16h b2 = *(const v16h*)(xb + 512 + 32 * c);
        v16h b3 = *(const v16h*)(xb + 768 + 32 * c);
        acc0 = __builtin_amdgcn_wmma_f32_16x16x32_f16(false, a, false, b0, (short)0, acc0, false, false);
        acc1 = __builtin_amdgcn_wmma_f32_16x16x32_f16(false, a, false, b1, (short)0, acc1, false, false);
        acc2 = __builtin_amdgcn_wmma_f32_16x16x32_f16(false, a, false, b2, (short)0, acc2, false, false);
        acc3 = __builtin_amdgcn_wmma_f32_16x16x32_f16(false, a, false, b3, (short)0, acc3, false, false);
    }
    cno_epilogue(acc0, i0,       lane, b, voiced, noise, out);
    cno_epilogue(acc1, i0 + 256, lane, b, voiced, noise, out);
    cno_epilogue(acc2, i0 + 512, lane, b, voiced, noise, out);
    cno_epilogue(acc3, i0 + 768, lane, b, voiced, noise, out);
}

// ---------------- launch ----------------
extern "C" void kernel_launch(void* const* d_in, const int* in_sizes, int n_in,
                              void* d_out, int out_size, void* d_ws, size_t ws_size,
                              hipStream_t stream) {
    const float* f0    = (const float*)d_in[0];   // [32,1,750]
    const float* noise = (const float*)d_in[1];   // [32,1,360000]
    const float* kern  = (const float*)d_in[2];   // [1,1,1003]
    float* out = (float*)d_out;

    // workspace layout
    char* ws = (char*)d_ws;
    const size_t AFRAG_OFF  = 0;                                   // 32768 B
    const size_t XBUF_OFF   = 65536;                               // 32*XSTRIDE*2 = 23,138,304 B
    const size_t VOICED_OFF = XBUF_OFF + (size_t)NBATCH * XSTRIDE * 2;      // 11,520,000 B
    const size_t PART_OFF   = VOICED_OFF + (size_t)NBATCH * L_SAMP;         // doubles
    const size_t NEEDED     = PART_OFF + (size_t)NBATCH * (2 * NCHUNK + 1) * 8 + 64;
    if (ws_size < NEEDED) return;   // cannot run without scratch

    _Float16* Afrag        = (_Float16*)(ws + AFRAG_OFF);
    _Float16* xbuf         = (_Float16*)(ws + XBUF_OFF);
    unsigned char* voiced  = (unsigned char*)(ws + VOICED_OFF);
    double* partial        = (double*)(ws + PART_OFF);
    double* prefix         = partial + (size_t)NBATCH * NCHUNK;

    cno_build_A<<<dim3(KCHUNKS), 32, 0, stream>>>(kern, Afrag);
    cno_chunk_sums<<<dim3(NCHUNK, NBATCH), SCANT, 0, stream>>>(f0, partial);
    cno_scan_chunks<<<dim3(1), 32, 0, stream>>>(partial, prefix);
    cno_gen_impulse<<<dim3(NCHUNK, NBATCH), SCANT, 0, stream>>>(f0, prefix, xbuf, voiced);
    cno_fir_wmma<<<dim3(QUADS / 4, NBATCH), 128, 0, stream>>>(xbuf, Afrag, voiced, noise, out);
}
